// GRAM_46832323396286
// MI455X (gfx1250) — matrix operand order repaired and checked by reading, not compiled, not run on hardware
//
#include <hip/hip_runtime.h>
#include <hip/hip_bf16.h>
#include <math.h>

typedef __attribute__((ext_vector_type(16))) _Float16 v16h;
typedef __attribute__((ext_vector_type(8)))  _Float16 v8h;
typedef __attribute__((ext_vector_type(8)))  float    v8f;

#define B_    32
#define V_    48
#define C_    24
#define A_    6
#define D_    128
#define H_    128
#define OUT_  167
#define R_    (C_*A_)     /* 144 cat rows per (b,v): 9 x 16 WMMA tiles */
#define TWO_D 256
#define NT    8           /* 8 N-tiles of 16 -> 128 */
#define KT    8           /* 8 K-steps of 32 -> 256 */
#define CATP  264         /* padded cat row stride in halfs: 528B = 132 dwords
                             -> consecutive rows 4 banks apart, 16B aligned */

// Branch-free transcendentals (map to v_exp_f32 / v_rcp_f32, no exec-divergent
// libm expansion in the WMMA epilogue).
__device__ __forceinline__ float fast_tanh(float x) {
  float t = __expf(2.0f * x);          // inf for large x, 0 for very negative
  return 1.0f - 2.0f / (t + 1.0f);     // -> +1 / -1 saturation, no NaN
}
__device__ __forceinline__ float fast_sigmoid(float x) {
  return 1.0f / (1.0f + __expf(-x));
}

// ---------------------------------------------------------------------------
// Kernel 0: weight prep.
//  - w_basic [128(e=N), 256(d=K)] f32 -> Wp2 f16, fragment-contiguous:
//      half index = (((kt*2 + hi)*128 + n)*8 + v)*2 + lo
//      with k = kt*32 + hi*16 + 2v + lo.  A lane's whole 16x16x32 B fragment
//      is then 8 consecutive dwords -> two global_load_b128.
//  - gru_wih/whh [384,128] -> transposed [128,384] for coalesced GRU reads.
// ---------------------------------------------------------------------------
__global__ void pack_weights(const float* __restrict__ w_basic,
                             const float* __restrict__ wih,
                             const float* __restrict__ whh,
                             _Float16* __restrict__ Wp2,
                             float* __restrict__ WihT,
                             float* __restrict__ WhhT) {
  int i = blockIdx.x * blockDim.x + threadIdx.x;
  if (i < D_ * TWO_D) {
    int k  = i >> 7;           // 0..255 (K)
    int n  = i & 127;          // 0..127 (N)
    int kt = k >> 5;
    int hi = (k >> 4) & 1;
    int v  = (k >> 1) & 7;
    int lo = k & 1;
    Wp2[(((kt * 2 + hi) * D_ + n) * 8 + v) * 2 + lo] = (_Float16)w_basic[n * TWO_D + k];
  }
  int j = i - D_ * TWO_D;
  if (j >= 0 && j < 384 * 128) {
    int r = j >> 7, k = j & 127;
    WihT[k * 384 + r] = wih[r * 128 + k];
  }
  int l = i - D_ * TWO_D - 384 * 128;
  if (l >= 0 && l < 384 * 128) {
    int r = l >> 7, k = l & 127;
    WhhT[k * 384 + r] = whh[r * 128 + k];
  }
}

// ---------------------------------------------------------------------------
// Kernel 1: per-(b,v) attention block, 9 waves (288 threads), dynamic LDS.
// Masked cat matrix staged once in f16, padded row-major [144][264]; A
// fragments are two conflict-free ds_load_b128 per K-step (A layout:
// K = hi*16+0..15, contiguous).  B fragments are two
// global_load_b128 from the fragment-contiguous f16 weight copy (L2-resident).
// GEMM [144x256]x[256x128] via v_wmma_f32_16x16x32_f16, K fully unrolled
// (64 WMMA).  Epilogue: tanh -> score dot (C-fragment layout + 16-lane
// butterfly) -> masked softmax over A -> ctx from masked anc half
// (equivalent since attn==0 wherever mask==0) -> x = tanh(sum_c ctx).
// ---------------------------------------------------------------------------
__global__ void __launch_bounds__(288)
gram_attention(const int*   __restrict__ seqs,
               const int*   __restrict__ ancestors,
               const float* __restrict__ amask,
               const float* __restrict__ emb,
               const uint4* __restrict__ Wp4,
               const float* __restrict__ u_w,
               const float* __restrict__ u_b,
               float* __restrict__ Xout) {
  extern __shared__ char smem[];
  _Float16* s_cat  = (_Float16*)smem;               // 144*264*2 = 76032 B
  float*    s_mask = (float*)(smem + 76032);        // 576 B
  float*    s_score= (float*)(smem + 76608);        // 576 B
  float*    s_attn = (float*)(smem + 77184);        // 576 B
  int*      s_sidx = (int*)  (smem + 77760);        // 96 B
  int*      s_aidx = (int*)  (smem + 77856);        // 576 B  (total 78432)

  const int tid = threadIdx.x;
  const int bv  = blockIdx.x;          // b*V + v

  if (tid < C_) s_sidx[tid] = seqs[bv * C_ + tid];
  if (tid < R_) {
    s_aidx[tid] = ancestors[bv * R_ + tid];
    s_mask[tid] = amask[bv * R_ + tid];
  }
  __syncthreads();

  // Stage masked cat = [seq_emb | anc_emb] * mask as f16, two halfs per u32
  // LDS store (k even; the 128-boundary is even so pairs never straddle it).
  for (int e = tid * 2; e < R_ * TWO_D; e += 288 * 2) {
    int r = e >> 8, k = e & 255;
    float mv = s_mask[r];
    const float* src = (k < D_) ? (emb + s_sidx[r / A_] * D_ + k)
                                : (emb + s_aidx[r] * D_ + (k - D_));
    union { unsigned u; _Float16 h[2]; } pk;
    pk.h[0] = (_Float16)(src[0] * mv);
    pk.h[1] = (_Float16)(src[1] * mv);
    *(unsigned*)(s_cat + r * CATP + k) = pk.u;
  }
  __syncthreads();

  const int wave = tid >> 5, lane = tid & 31;
  const int col = lane & 15, hi = lane >> 4;
  const int m0 = wave * 16;
  const int arow = m0 + col;                     // this lane's A-fragment row
  const _Float16* arowp = s_cat + arow * CATP;

  v8f acc[NT];
  #pragma unroll
  for (int n = 0; n < NT; ++n)
    #pragma unroll
    for (int r = 0; r < 8; ++r) acc[n][r] = 0.0f;

  #pragma unroll
  for (int kt = 0; kt < KT; ++kt) {
    const int koff = kt * 32;
    // A fragment: two contiguous 8-half runs -> two ds_load_b128.
    union { v16h v; v8h h[2]; } au;
    au.h[0] = *(const v8h*)(arowp + koff + hi * 16);
    au.h[1] = *(const v8h*)(arowp + koff + hi * 16 + 8);
    // Base (in uint4 units) of this lane's fragment-contiguous B data.
    const uint4* wbase = Wp4 + ((kt * 2 + hi) * D_ + col) * 2;
    #pragma unroll
    for (int nt = 0; nt < NT; ++nt) {
      union { v16h v; uint4 q[2]; } bu;
      bu.q[0] = wbase[nt * 32];        // v = 0..3  (16 halfs consecutive)
      bu.q[1] = wbase[nt * 32 + 1];    // v = 4..7
      acc[nt] = __builtin_amdgcn_wmma_f32_16x16x32_f16(
          false, au.v, false, bu.v, (short)0, acc[nt], false, false);
    }
  }

  // Epilogue: hdd = tanh(acc); score_row = dot(hdd_row, u_w).
  float uu[NT];
  #pragma unroll
  for (int nt = 0; nt < NT; ++nt) uu[nt] = u_w[nt * 16 + col];
  float p[8];
  #pragma unroll
  for (int r = 0; r < 8; ++r) p[r] = 0.0f;
  #pragma unroll
  for (int nt = 0; nt < NT; ++nt)
    #pragma unroll
    for (int r = 0; r < 8; ++r)
      p[r] += fast_tanh(acc[nt][r]) * uu[nt];
  // Reduce the 16 columns held across each 16-lane half-wave.
  #pragma unroll
  for (int r = 0; r < 8; ++r)
    for (int m = 1; m < 16; m <<= 1)
      p[r] += __shfl_xor(p[r], m, 32);
  if (col == 0) {
    float ub = u_b[0];
    #pragma unroll
    for (int r = 0; r < 8; ++r)
      s_score[m0 + hi * 8 + r] = p[r] + ub;      // C layout: row M = r + 8*hi
  }
  __syncthreads();

  // Masked softmax over ancestors (reference: sum==0 -> -1).
  if (tid < C_) {
    float ev[A_]; float s = 0.0f;
    #pragma unroll
    for (int a = 0; a < A_; ++a) {
      ev[a] = __expf(s_score[tid * A_ + a]) * s_mask[tid * A_ + a];
      s += ev[a];
    }
    if (s == 0.0f) s = -1.0f;
    #pragma unroll
    for (int a = 0; a < A_; ++a) s_attn[tid * A_ + a] = ev[a] / s;
  }
  __syncthreads();

  // x[d] = tanh( sum_{c,a} attn * anc_e[d] ); masked anc half is equivalent
  // because attn==0 wherever mask==0.
  if (tid < D_) {
    float xs = 0.0f;
    for (int r = 0; r < R_; ++r)
      xs += s_attn[r] * (float)s_cat[r * CATP + D_ + tid];
    Xout[bv * D_ + tid] = fast_tanh(xs);
  }
}

// ---------------------------------------------------------------------------
// Kernel 2: per-batch GRU (independent recurrences), fused masked visit sum
// and sigmoid output head.  Thread j owns h[j]; weights transposed for
// coalesced reads (L2-resident, 384 KB total).
// ---------------------------------------------------------------------------
__global__ void __launch_bounds__(128)
gram_gru(const float* __restrict__ X, const int* __restrict__ length,
         const float* __restrict__ WihT, const float* __restrict__ WhhT,
         const float* __restrict__ bih, const float* __restrict__ bhh,
         const float* __restrict__ out_w, const float* __restrict__ out_b,
         float* __restrict__ out) {
  __shared__ float sh_x[D_];
  __shared__ float sh_h[H_];
  __shared__ float s_ctx[H_];
  const int b = blockIdx.x;
  const int j = threadIdx.x;
  const int len = length[b];
  sh_h[j] = 0.0f;
  float ctx = 0.0f;
  const float br = bih[j], bz = bih[H_ + j], bn = bih[2 * H_ + j];
  const float cr = bhh[j], cz = bhh[H_ + j], cn = bhh[2 * H_ + j];
  __syncthreads();

  for (int t = 0; t < V_; ++t) {
    sh_x[j] = X[(b * V_ + t) * D_ + j];
    __syncthreads();
    float gir = br, giz = bz, gin = bn, ghr = cr, ghz = cz, ghn = cn;
    for (int k = 0; k < D_; ++k) {
      const float xv = sh_x[k], hv = sh_h[k];
      const float* wi = WihT + k * 384;
      const float* wh = WhhT + k * 384;
      gir += wi[j] * xv;            ghr += wh[j] * hv;
      giz += wi[H_ + j] * xv;       ghz += wh[H_ + j] * hv;
      gin += wi[2 * H_ + j] * xv;   ghn += wh[2 * H_ + j] * hv;
    }
    const float r = fast_sigmoid(gir + ghr);
    const float z = fast_sigmoid(giz + ghz);
    const float n = fast_tanh(gin + r * ghn);
    const float hnew = (1.0f - z) * n + z * sh_h[j];
    __syncthreads();
    sh_h[j] = hnew;
    __syncthreads();
    if (t < len) ctx += hnew;       // masked visit sum
  }
  s_ctx[j] = ctx;
  __syncthreads();
  for (int o = j; o < OUT_; o += H_) {
    float a = out_b[o];
    for (int k = 0; k < H_; ++k) a += s_ctx[k] * out_w[o * H_ + k];
    out[b * OUT_ + o] = fast_sigmoid(a);
  }
}

// ---------------------------------------------------------------------------
extern "C" void kernel_launch(void* const* d_in, const int* in_sizes, int n_in,
                              void* d_out, int out_size, void* d_ws, size_t ws_size,
                              hipStream_t stream) {
  const int*   seqs      = (const int*)  d_in[0];
  const int*   ancestors = (const int*)  d_in[1];
  const int*   length    = (const int*)  d_in[2];
  const float* amask     = (const float*)d_in[3];
  const float* emb       = (const float*)d_in[4];
  const float* w_basic   = (const float*)d_in[5];
  const float* u_w       = (const float*)d_in[6];
  const float* u_b       = (const float*)d_in[7];
  const float* gru_wih   = (const float*)d_in[8];
  const float* gru_whh   = (const float*)d_in[9];
  const float* gru_bih   = (const float*)d_in[10];
  const float* gru_bhh   = (const float*)d_in[11];
  const float* out_w     = (const float*)d_in[12];
  const float* out_b     = (const float*)d_in[13];
  float* out = (float*)d_out;

  // Workspace layout (all 256B-aligned): ~1.25 MB total.
  char* ws = (char*)d_ws;
  _Float16* Wp2  = (_Float16*)ws;                       //  65536 B
  float*    WihT = (float*)(ws + 65536);                // 196608 B
  float*    WhhT = (float*)(ws + 65536 + 196608);       // 196608 B
  float*    X    = (float*)(ws + 65536 + 2 * 196608);   // 786432 B

  const int attn_lds = 78432;   // ~76.6 KB dynamic LDS (gfx1250: 320 KB/WGP)
  (void)hipFuncSetAttribute((const void*)gram_attention,
                            hipFuncAttributeMaxDynamicSharedMemorySize,
                            attn_lds);

  pack_weights<<<512, 256, 0, stream>>>(w_basic, gru_wih, gru_whh, Wp2, WihT, WhhT);
  gram_attention<<<B_ * V_, 288, attn_lds, stream>>>(seqs, ancestors, amask, emb,
                                                     (const uint4*)Wp2, u_w, u_b, X);
  gram_gru<<<B_, 128, 0, stream>>>(X, length, WihT, WhhT, gru_bih, gru_bhh,
                                   out_w, out_b, out);
}
